// AE_spikes_72198400245922
// MI455X (gfx1250) — compile-verified, matrix operand order checked by reading, not run on hardware
//
#include <hip/hip_runtime.h>
#include <hip/hip_bf16.h>

typedef __attribute__((ext_vector_type(16))) _Float16 v16h;
typedef __attribute__((ext_vector_type(8)))  float    v8f;

#define B_TOTAL 16384
#define IN_DIM  784
#define INP     800      // K padded to multiple of 32 for layer 1
#define INPS    808      // LDS row stride for s0 masks (404 dwords -> bank rotation 20)
#define HID     128
#define HROW    136      // LDS row stride for s1/s2/s3m (68 dwords -> bank rotation 4)
#define TSTEPS  16
#define BT      64       // batch rows per workgroup
#define NKC1    25       // INP/32
#define NKCH    4        // HID/32
#define NCBH    8        // HID/16
#define NCBO    49       // 784/16

// workspace layout (units: halves). Each fragment = 32 lanes * 16 halves = 512.
#define W1F_OFF 0
#define W1F_LEN (NKC1*NCBH*512)   // 102400
#define W2F_OFF (W1F_OFF + W1F_LEN)
#define W2F_LEN (NKCH*NCBH*512)   // 16384
#define W3F_OFF (W2F_OFF + W2F_LEN)
#define W3F_LEN (NKCH*NCBH*512)
#define W4F_OFF (W3F_OFF + W3F_LEN)
#define W4F_LEN (NKCH*NCBO*512)   // 100352
#define WSF_TOTAL (W4F_OFF + W4F_LEN)

// LDS: s0 masks [64][808]u16 + s1/s2 spikes [64][136]u16 + s3 masks [64][136]u16
#define S0M_BYTES (BT*INPS*2)          // 103424
#define SXX_BYTES (BT*HROW*2)          // 17408
#define SMEM_BYTES (S0M_BYTES + 3*SXX_BYTES)   // 155648

union VH { v16h h; uint4 u[2]; unsigned d[8]; };

struct MaskRaw { uint4 lo, hi; };

__device__ __forceinline__ v8f wmma16(v16h a, v16h b, v8f c) {
  // D = A(16x32 f16) * B(32x16 f16) + C(16x16 f32)
  return __builtin_amdgcn_wmma_f32_16x16x32_f16(false, a, false, b, (short)0, c,
                                                false, false);
}

// Load a pre-swizzled B fragment: 32 bytes per lane (two global_load_b128).
__device__ __forceinline__ v16h load_bfrag(const _Float16* __restrict__ base,
                                           int fragIdx, int lane) {
  const uint4* p = (const uint4*)(base + (((size_t)fragIdx * 32 + lane) << 4));
  VH r; r.u[0] = p[0]; r.u[1] = p[1];
  return r.h;
}

// Build f16 A fragment from a row-major u16-spike LDS buffer (stride halves).
// Lane layout (16-bit A 16x32): row = lane&15; laneHalf (lane>>4) selects
// K = 8*half + {0..7} (dwords 0-3) and K = 16 + 8*half + {0..7} (dwords 4-7).
__device__ __forceinline__ v16h build_a_f16(const unsigned short* sbuf, int stride,
                                            int rbase, int lane, int kc) {
  int r  = rbase + (lane & 15);
  int kh = (lane >> 4) << 3;
  const uint4* q = (const uint4*)(sbuf + r * stride + kc * 32 + kh);
  VH a; a.u[0] = q[0]; a.u[1] = q[2];   // +32B == +16 halves
  return a.h;
}

// Raw 16-step bitmask dwords for one A fragment (t-invariant part).
__device__ __forceinline__ MaskRaw load_mask_raw(const unsigned short* mbuf, int stride,
                                                 int rbase, int lane, int kc) {
  int r  = rbase + (lane & 15);
  int kh = (lane >> 4) << 3;
  const uint4* q = (const uint4*)(mbuf + r * stride + kc * 32 + kh);
  MaskRaw m; m.lo = q[0]; m.hi = q[2];
  return m;
}

// Expand step-t spikes to f16 {0,1}: ((m >> t) & 0x00010001) * 0x3C00.
__device__ __forceinline__ v16h expand_mask(const MaskRaw& m, int t) {
  const unsigned* lp = (const unsigned*)&m.lo;
  const unsigned* hp = (const unsigned*)&m.hi;
  VH a;
#pragma unroll
  for (int d = 0; d < 4; ++d) {
    a.d[d]     = (unsigned)__mul24((int)((lp[d] >> t) & 0x00010001u), 0x3C00);
    a.d[4 + d] = (unsigned)__mul24((int)((hp[d] >> t) & 0x00010001u), 0x3C00);
  }
  return a.h;
}

// Hidden-layer GEMM tile (K=128): explicit double-buffered B + A look-ahead.
__device__ __forceinline__ void layer_hid(const unsigned short* sbuf,
                                          const _Float16* __restrict__ Wf,
                                          int rb, int cb0, int lane, v8f (&acc)[4]) {
  v16h a = build_a_f16(sbuf, HROW, rb * 16, lane, 0);
  v16h bb[2][4];
#pragma unroll
  for (int j = 0; j < 4; ++j) bb[0][j] = load_bfrag(Wf, cb0 + j, lane);
#pragma unroll
  for (int kc = 0; kc < NKCH; ++kc) {
    if (kc + 1 < NKCH) {
      v16h an = build_a_f16(sbuf, HROW, rb * 16, lane, kc + 1);
#pragma unroll
      for (int j = 0; j < 4; ++j)
        bb[(kc + 1) & 1][j] = load_bfrag(Wf, (kc + 1) * NCBH + cb0 + j, lane);
#pragma unroll
      for (int j = 0; j < 4; ++j) acc[j] = wmma16(a, bb[kc & 1][j], acc[j]);
      a = an;
    } else {
#pragma unroll
      for (int j = 0; j < 4; ++j) acc[j] = wmma16(a, bb[kc & 1][j], acc[j]);
    }
  }
}

// Pre-swizzle weights (row-major [N][Kdim] f32, element(K,N)=W[N*Kdim+K]) into
// fragment-major f16: dst[((kc*ncb+cb)*32+lane)*16 + j*2+p], K per WMMA striping.
__global__ void prep_frags(const float* __restrict__ W, _Float16* __restrict__ dst,
                           int Kdim, int nkc, int ncb) {
  int idx = blockIdx.x * blockDim.x + threadIdx.x;
  int total = nkc * ncb * 512;
  if (idx >= total) return;
  int j2   = idx & 15;
  int lane = (idx >> 4) & 31;
  int fc   = idx >> 9;
  int cb   = fc % ncb;
  int kc   = fc / ncb;
  int j = j2 >> 1, p = j2 & 1;
  int kk = (j < 4 ? 2 * j : 16 + 2 * (j - 4)) + ((lane >> 4) << 3) + p;
  int K = kc * 32 + kk;
  int N = cb * 16 + (lane & 15);
  float v = (K < Kdim) ? W[(size_t)N * Kdim + K] : 0.0f;
  dst[idx] = (_Float16)v;
}

extern "C" __global__ __launch_bounds__(256)
__attribute__((amdgpu_num_vgpr(256)))
void snn_ae_main(const float* __restrict__ features,
                 const _Float16* __restrict__ wf,
                 const float* __restrict__ out_scale,
                 float* __restrict__ out) {
  extern __shared__ __align__(16) char smem[];
  unsigned short* s0m = (unsigned short*)smem;                     // [BT][INPS]
  unsigned short* s1  = (unsigned short*)(smem + S0M_BYTES);       // [BT][HROW]
  unsigned short* s2  = s1 + BT * HROW;                            // [BT][HROW]
  unsigned short* s3m = s2 + BT * HROW;                            // [BT][HROW]

  const int tid  = threadIdx.x;
  const int lane = tid & 31;
  const int w    = tid >> 5;           // wave 0..7
  const int b0   = blockIdx.x * BT;

  const _Float16* W1f = wf + W1F_OFF;
  const _Float16* W2f = wf + W2F_OFF;
  const _Float16* W3f = wf + W3F_OFF;
  const _Float16* W4f = wf + W4F_OFF;

  // ---- Phase 0: rate-encode inputs into 16-step spike bitmasks (p0 closed form)
  for (int e = tid; e < BT * INPS; e += 256) {
    int r = e / INPS, k = e - r * INPS;
    unsigned m = 0;
    if (k < IN_DIM) {
      float f = features[(size_t)(b0 + r) * IN_DIM + k];
      int L = (int)roundf(f * 16.0f);  // 0..16 ; spikes(t) = floor((t+1)L/16)-floor(tL/16)
      int prev = 0;
#pragma unroll
      for (int t = 1; t <= TSTEPS; ++t) {
        int cur = (t * L) >> 4;
        m |= (unsigned)(cur - prev) << (t - 1);
        prev = cur;
      }
    }
    s0m[e] = (unsigned short)m;
  }
  __syncthreads();

  // ---- Phase A: per-step layers 1..3, potentials persistent in VGPRs.
  // wave owns 1 rowblock (w&3) and 4 colblocks ((w>>2)*4..): one A build
  // feeds 4 WMMAs per k-chunk.
  const int rb  = w & 3;
  const int cb0 = (w >> 2) * 4;
  const int rr  = rb * 16 + ((lane >> 4) << 3);     // C-layout row base for this lane
  const int cc  = cb0 * 16 + (lane & 15);           // C-layout col base for this lane

  v8f p1[4] = {};
  v8f p2[4] = {};
  v8f p3[4] = {};
  unsigned m3p[4][4] = {};   // packed: two 16-bit time masks per dword

#pragma unroll 1
  for (int t = 0; t < TSTEPS; ++t) {
    // ----- Layer 1: 800 -> 128, A from input spike masks.
    // Explicit 2-deep pipeline: B frags + raw masks for kc+1 in flight while
    // WMMAs for kc execute (no buffer-rotation copies: step-2 loop + tail).
    {
      MaskRaw r0 = load_mask_raw(s0m, INPS, rb * 16, lane, 0);
      v16h b0[4], b1[4];
#pragma unroll
      for (int j = 0; j < 4; ++j) b0[j] = load_bfrag(W1f, cb0 + j, lane);
#pragma unroll 1
      for (int kc = 0; kc < NKC1 - 1; kc += 2) {
        MaskRaw r1 = load_mask_raw(s0m, INPS, rb * 16, lane, kc + 1);
#pragma unroll
        for (int j = 0; j < 4; ++j)
          b1[j] = load_bfrag(W1f, (kc + 1) * NCBH + cb0 + j, lane);
        v16h a = expand_mask(r0, t);
#pragma unroll
        for (int j = 0; j < 4; ++j) p1[j] = wmma16(a, b0[j], p1[j]);
        r0 = load_mask_raw(s0m, INPS, rb * 16, lane, kc + 2);
#pragma unroll
        for (int j = 0; j < 4; ++j)
          b0[j] = load_bfrag(W1f, (kc + 2) * NCBH + cb0 + j, lane);
        a = expand_mask(r1, t);
#pragma unroll
        for (int j = 0; j < 4; ++j) p1[j] = wmma16(a, b1[j], p1[j]);
      }
      v16h a = expand_mask(r0, t);      // kc = 24 tail
#pragma unroll
      for (int j = 0; j < 4; ++j) p1[j] = wmma16(a, b0[j], p1[j]);
    }
    // fire layer-1, spikes -> s1 LDS (C layout: row = rr+g, col = cc+16j)
#pragma unroll
    for (int j = 0; j < 4; ++j)
#pragma unroll
      for (int g = 0; g < 8; ++g) {
        bool fr = (p1[j][g] >= 1.0f);
        p1[j][g] -= fr ? 1.0f : 0.0f;
        s1[(rr + g) * HROW + cc + j * 16] = fr ? (unsigned short)0x3C00 : (unsigned short)0;
      }
    __syncthreads();

    // ----- Layer 2: 128 -> 128
    layer_hid(s1, W2f, rb, cb0, lane, p2);
#pragma unroll
    for (int j = 0; j < 4; ++j)
#pragma unroll
      for (int g = 0; g < 8; ++g) {
        bool fr = (p2[j][g] >= 1.0f);
        p2[j][g] -= fr ? 1.0f : 0.0f;
        s2[(rr + g) * HROW + cc + j * 16] = fr ? (unsigned short)0x3C00 : (unsigned short)0;
      }
    __syncthreads();

    // ----- Layer 3: 128 -> 128, spikes accumulated as packed time-bitmasks
    layer_hid(s2, W3f, rb, cb0, lane, p3);
#pragma unroll
    for (int j = 0; j < 4; ++j)
#pragma unroll
      for (int g = 0; g < 8; ++g) {
        unsigned bit = (p3[j][g] >= 1.0f) ? 1u : 0u;
        p3[j][g] -= (float)bit;
        m3p[j][g >> 1] |= bit << (t + ((g & 1) << 4));
      }
    // no barrier needed: next-iteration s1 stores are fenced by this step's barriers
  }

  // dump s3 time-bitmasks to LDS for phase B
#pragma unroll
  for (int j = 0; j < 4; ++j)
#pragma unroll
    for (int g = 0; g < 8; ++g)
      s3m[(rr + g) * HROW + cc + j * 16] =
          (unsigned short)(m3p[j][g >> 1] >> ((g & 1) << 4));
  __syncthreads();

  // ---- Phase B: decoder 128 -> 784; p4 + spike counts in registers.
  // W4 B-fragments AND raw s3 masks (t-invariant) cached in registers across
  // the 16 timesteps: the recurrent loop is pure VALU-expand + WMMA.
  // 50 groups = (rowblock-pair 0..1) x (25 groups of up-to-2 output colblocks)
  const float scl = out_scale[0] * (1.0f / 16.0f);
#pragma unroll 1
  for (int grp = w; grp < 50; grp += 8) {
    int rbp = grp / 25;
    int og  = grp % 25;
    int ob0 = og * 2;
    int nob = (ob0 + 2 <= NCBO) ? 2 : (NCBO - ob0);
    v16h bfr[NKCH][2];
    MaskRaw mr[NKCH][2];
#pragma unroll
    for (int kc = 0; kc < NKCH; ++kc) {
      mr[kc][0] = load_mask_raw(s3m, HROW, rbp * 32,      lane, kc);
      mr[kc][1] = load_mask_raw(s3m, HROW, rbp * 32 + 16, lane, kc);
#pragma unroll
      for (int j = 0; j < 2; ++j)
        if (j < nob) bfr[kc][j] = load_bfrag(W4f, kc * NCBO + ob0 + j, lane);
    }
    v8f p4[2][2]  = {};
    v8f cnt[2][2] = {};
#pragma unroll 1
    for (int t = 0; t < TSTEPS; ++t) {
#pragma unroll
      for (int kc = 0; kc < NKCH; ++kc) {
        v16h a0 = expand_mask(mr[kc][0], t);
        v16h a1 = expand_mask(mr[kc][1], t);
#pragma unroll
        for (int j = 0; j < 2; ++j)
          if (j < nob) {
            p4[0][j] = wmma16(a0, bfr[kc][j], p4[0][j]);
            p4[1][j] = wmma16(a1, bfr[kc][j], p4[1][j]);
          }
      }
#pragma unroll
      for (int i = 0; i < 2; ++i)
#pragma unroll
        for (int j = 0; j < 2; ++j)
          if (j < nob)
#pragma unroll
            for (int g = 0; g < 8; ++g) {
              float s = (p4[i][j][g] >= 1.0f) ? 1.0f : 0.0f;
              p4[i][j][g] -= s;
              cnt[i][j][g] += s;
            }
    }
#pragma unroll
    for (int i = 0; i < 2; ++i)
#pragma unroll
      for (int j = 0; j < 2; ++j)
        if (j < nob)
#pragma unroll
          for (int g = 0; g < 8; ++g) {
            int row = b0 + (rbp * 2 + i) * 16 + ((lane >> 4) << 3) + g;
            int col = (ob0 + j) * 16 + (lane & 15);
            out[(size_t)row * IN_DIM + col] = cnt[i][j][g] * scl;
          }
  }
}

extern "C" void kernel_launch(void* const* d_in, const int* in_sizes, int n_in,
                              void* d_out, int out_size, void* d_ws, size_t ws_size,
                              hipStream_t stream) {
  (void)in_sizes; (void)n_in; (void)out_size; (void)ws_size;
  const float* features  = (const float*)d_in[0];
  const float* W1        = (const float*)d_in[1];
  const float* W2        = (const float*)d_in[2];
  const float* W3        = (const float*)d_in[3];
  const float* W4        = (const float*)d_in[4];
  const float* out_scale = (const float*)d_in[5];
  _Float16* wf = (_Float16*)d_ws;   // needs WSF_TOTAL*2 = 471040 bytes

  dim3 blk(256);
  prep_frags<<<(W1F_LEN + 255) / 256, blk, 0, stream>>>(W1, wf + W1F_OFF, IN_DIM, NKC1, NCBH);
  prep_frags<<<(W2F_LEN + 255) / 256, blk, 0, stream>>>(W2, wf + W2F_OFF, HID,    NKCH, NCBH);
  prep_frags<<<(W3F_LEN + 255) / 256, blk, 0, stream>>>(W3, wf + W3F_OFF, HID,    NKCH, NCBH);
  prep_frags<<<(W4F_LEN + 255) / 256, blk, 0, stream>>>(W4, wf + W4F_OFF, HID,    NKCH, NCBO);

  (void)hipFuncSetAttribute(reinterpret_cast<const void*>(snn_ae_main),
                            hipFuncAttributeMaxDynamicSharedMemorySize, SMEM_BYTES);
  snn_ae_main<<<B_TOTAL / BT, blk, SMEM_BYTES, stream>>>(features, wf, out_scale,
                                                         (float*)d_out);
}
